// ContextModulatedNeighborhoodAttention_38680475468379
// MI455X (gfx1250) — compile-verified, matrix operand order checked by reading, not run on hardware
//
#include <hip/hip_runtime.h>
#include <hip/hip_bf16.h>
#include <math.h>

// ---------------------------------------------------------------------------
// ContextModulatedNeighborhoodAttention for MI455X (gfx1250, wave32, WMMA)
//
// Shapes: B=4, H=W=64, C=128, 3C=384, nh=4, hd=32, K=7 (pad 3), CTX=256, r=8
//
// LoRA folding:  qkv = x @ (Wqkv^T + alpha_b * A diag(cB_b) Vlora^T) + bqkv
// so the modulated projection is a plain per-batch GEMM against Weff_b.
//
// GEMM scheme per block (128 thr = 4 waves):
//   - block owns a 64-row x 64-col output panel
//   - 64x128 bf16 B panel staged once into LDS (async-to-LDS if available)
//   - each wave: 16 rows x 64 cols = 4 accumulators, 16 WMMAs, A fragment
//     reused across 4 N-tiles
// ---------------------------------------------------------------------------

typedef __attribute__((ext_vector_type(16))) __bf16 v16bf;
typedef __attribute__((ext_vector_type(8)))  float  v8f;
typedef __attribute__((ext_vector_type(4)))  int    v4i;

#define BATCH   4
#define HW      4096          // 64*64
#define EMBED   128
#define C3      384
#define NHEADS  4
#define HDIM    32
#define CTXD    256
#define RANKD   8
#define SCALEF  0.17677669529663687f   // 32^-0.5
#define BSTRIDE 136           // LDS row stride in bf16 (128 + 8 pad -> bank spread)

#if defined(__has_builtin)
#if __has_builtin(__builtin_amdgcn_global_load_async_to_lds_b128) && \
    __has_builtin(__builtin_amdgcn_s_wait_asynccnt)
#define USE_ASYNC_LDS 1
#endif
#endif

#ifdef USE_ASYNC_LDS
typedef __attribute__((address_space(1))) v4i as1_v4i;   // global (prints __device__)
typedef __attribute__((address_space(3))) v4i as3_v4i;   // LDS    (prints __shared__)
#endif

// Stage a 64-col x 128-K bf16 panel (B matrix, [col][k] layout) into LDS.
// 128 threads: thread t copies col = t/2, k-half = (t&1)*64 (128 B = 8 x b128).
__device__ __forceinline__ void stage_b_panel(const __bf16* __restrict__ gsrc,
                                              __bf16* __restrict__ ldsB) {
  const int t   = threadIdx.x;
  const int col = t >> 1;
  const int ko  = (t & 1) * 64;
  const __bf16* src = gsrc + (size_t)col * EMBED + ko;
  __bf16*       dst = ldsB + col * BSTRIDE + ko;
#ifdef USE_ASYNC_LDS
#pragma unroll
  for (int c = 0; c < 8; ++c) {
    __builtin_amdgcn_global_load_async_to_lds_b128(
        (as1_v4i*)(src + c * 8), (as3_v4i*)(dst + c * 8), 0, 0);
  }
  __builtin_amdgcn_s_wait_asynccnt(0);
#else
#pragma unroll
  for (int c = 0; c < 8; ++c) {
    *(uint4*)(dst + c * 8) = *(const uint4*)(src + c * 8);
  }
#endif
  __syncthreads();
}

// ---------------------------------------------------------------------------
// Kernel 0: per-batch scalars.  c_proj[b][r] = context[b] @ Blora[:,r]
//           alpha[b] = sigmoid( relu(ctx@g1w^T+g1b) @ g2w^T + g2b )
// ---------------------------------------------------------------------------
__global__ __launch_bounds__(64) void prep_scalars(
    const float* __restrict__ context, const float* __restrict__ Blora,
    const float* __restrict__ g1w, const float* __restrict__ g1b,
    const float* __restrict__ g2w, const float* __restrict__ g2b,
    float* __restrict__ c_proj, float* __restrict__ alpha) {
  const int t = threadIdx.x;
  if (t < BATCH * RANKD) {
    const int b = t >> 3, r = t & 7;
    float s = 0.0f;
    for (int i = 0; i < CTXD; ++i) s = fmaf(context[b * CTXD + i], Blora[i * RANKD + r], s);
    c_proj[t] = s;
  } else if (t < BATCH * RANKD + BATCH) {
    const int b = t - BATCH * RANKD;
    float acc = g2b[0];
    for (int j = 0; j < 16; ++j) {
      float hsum = g1b[j];
      for (int i = 0; i < CTXD; ++i) hsum = fmaf(context[b * CTXD + i], g1w[j * CTXD + i], hsum);
      hsum = fmaxf(hsum, 0.0f);
      acc = fmaf(hsum, g2w[j], acc);
    }
    alpha[b] = 1.0f / (1.0f + __expf(-acc));
  }
}

// ---------------------------------------------------------------------------
// Kernel 1: effective weights.
//  Weff[b][n][k] = Wqkv[n][k] + alpha[b] * sum_r A[k][r]*c_proj[b][r]*Vlora[n][r]
//  ([feature n][input k] layout -> B-panel rows are contiguous along K)
//  WprojBf[n][k] = bf16(Wproj[n][k])
// ---------------------------------------------------------------------------
#define WEFF_ELEMS (BATCH * C3 * EMBED)     // 196608
#define WPROJ_ELEMS (EMBED * EMBED)         // 16384

__global__ __launch_bounds__(256) void prep_weights(
    const float* __restrict__ Wqkv, const float* __restrict__ A,
    const float* __restrict__ Vlora, const float* __restrict__ c_proj,
    const float* __restrict__ alpha, const float* __restrict__ Wproj,
    __bf16* __restrict__ Weff, __bf16* __restrict__ WprojBf) {
  const int idx = blockIdx.x * 256 + threadIdx.x;
  if (idx < WEFF_ELEMS) {
    const int b = idx / (C3 * EMBED);
    const int rem = idx - b * (C3 * EMBED);
    const int n = rem >> 7;          // /128
    const int k = rem & 127;
    float s = 0.0f;
#pragma unroll
    for (int r = 0; r < RANKD; ++r)
      s = fmaf(A[k * RANKD + r] * c_proj[b * RANKD + r], Vlora[n * RANKD + r], s);
    Weff[idx] = (__bf16)(Wqkv[n * EMBED + k] + alpha[b] * s);
  } else if (idx < WEFF_ELEMS + WPROJ_ELEMS) {
    const int j = idx - WEFF_ELEMS;
    WprojBf[j] = (__bf16)Wproj[j];
  }
}

// ---------------------------------------------------------------------------
// WMMA fragment mapping (ISA 7.12.2, wave32):
//  A (16x32 bf16): lane%16 = M row; element e -> K = (e<8?0:16) + (lane>=16?8:0) + (e&7)
//  B (32x16 bf16): lane%16 = N col; same K mapping (mirrored)
//  C/D (16x16 f32): element v -> M = v + 8*(lane/16); N = lane%16
// ---------------------------------------------------------------------------

// Kernel 2: qkv = x @ Weff_b^T + bqkv.  Per batch: M=4096, N=384, K=128.
// grid(M/64, N/64, B), block 128.  Wave w -> rows (bx*64 + w*16 .. +16).
__global__ __launch_bounds__(128) void qkv_gemm(
    const float* __restrict__ x, const __bf16* __restrict__ Weff,
    const float* __restrict__ bqkv, float* __restrict__ qkv) {
  __shared__ __bf16 ldsB[64 * BSTRIDE];

  const int lane = threadIdx.x & 31;
  const int wave = threadIdx.x >> 5;
  const int lm   = lane & 15;
  const int half = lane >> 4;
  const int h8   = half * 8;
  const int b    = blockIdx.z;
  const int row0 = blockIdx.x * 64 + wave * 16;
  const int col0 = blockIdx.y * 64;

  stage_b_panel(Weff + (size_t)b * C3 * EMBED + (size_t)col0 * EMBED, ldsB);

  const float* arow = x + ((size_t)(b * HW + row0 + lm)) * EMBED;

  v8f acc0 = {}, acc1 = {}, acc2 = {}, acc3 = {};
#pragma unroll
  for (int k0 = 0; k0 < EMBED; k0 += 32) {
    // A fragment: f32 -> bf16, reused across 4 N-tiles.
    v16bf af;
    {
      const float* ap = arow + k0 + h8;
#pragma unroll
      for (int e = 0; e < 8; ++e) {
        af[e]     = (__bf16)ap[e];
        af[e + 8] = (__bf16)ap[e + 16];
      }
    }
#pragma unroll
    for (int nt = 0; nt < 4; ++nt) {
      v16bf bfr;
      const __bf16* bp = ldsB + (nt * 16 + lm) * BSTRIDE + k0 + h8;
#pragma unroll
      for (int e = 0; e < 8; ++e) {
        bfr[e]     = bp[e];
        bfr[e + 8] = bp[e + 16];
      }
      v8f* accp = (nt == 0) ? &acc0 : (nt == 1) ? &acc1 : (nt == 2) ? &acc2 : &acc3;
      *accp = __builtin_amdgcn_wmma_f32_16x16x32_bf16(false, af, false, bfr,
                                                      (short)0, *accp, false, false);
    }
  }
#pragma unroll
  for (int nt = 0; nt < 4; ++nt) {
    const v8f acc = (nt == 0) ? acc0 : (nt == 1) ? acc1 : (nt == 2) ? acc2 : acc3;
    const int colT = col0 + nt * 16;
    const float bias = bqkv[colT + lm];
    float* orow = qkv + ((size_t)(b * HW + row0 + 8 * half)) * C3 + colT + lm;
#pragma unroll
    for (int v = 0; v < 8; ++v) orow[(size_t)v * C3] = acc[v] + bias;
  }
}

// ---------------------------------------------------------------------------
// Kernel 3: 7x7 neighborhood attention.  One thread = one (b, head, y, x).
// qkv layout: [b][pixel][384] with q at +0, k at +128, v at +256 (+h*32).
// Zero-padded OOB taps contribute logit 0 to softmax (matches jnp.pad ref).
// ---------------------------------------------------------------------------
__global__ __launch_bounds__(256) void neigh_attn(
    const float* __restrict__ qkv, __bf16* __restrict__ attn_out) {
  const int idx = blockIdx.x * 256 + threadIdx.x;   // 65536 total
  const int x = idx & 63;
  const int y = (idx >> 6) & 63;
  const int h = (idx >> 12) & 3;
  const int b = idx >> 14;
  const size_t pixBase = (size_t)(b * HW + y * 64 + x) * C3;

  float logit[49];
  {
    float q[HDIM];
    const float* qp = qkv + pixBase + h * HDIM;
#pragma unroll
    for (int d = 0; d < HDIM; ++d) q[d] = qp[d];
#pragma unroll
    for (int ky = 0; ky < 7; ++ky) {
#pragma unroll
      for (int kx = 0; kx < 7; ++kx) {
        const int yy = y + ky - 3;
        const int xx = x + kx - 3;
        float s = 0.0f;
        if ((unsigned)yy < 64u && (unsigned)xx < 64u) {
          const float* kp = qkv + (size_t)(b * HW + yy * 64 + xx) * C3 + EMBED + h * HDIM;
#pragma unroll
          for (int d = 0; d < HDIM; ++d) s = fmaf(q[d], kp[d], s);
          s *= SCALEF;
        }
        logit[ky * 7 + kx] = s;
      }
    }
  }
  float mx = logit[0];
#pragma unroll
  for (int wi = 1; wi < 49; ++wi) mx = fmaxf(mx, logit[wi]);
  float den = 0.0f;
#pragma unroll
  for (int wi = 0; wi < 49; ++wi) {
    const float e = __expf(logit[wi] - mx);
    logit[wi] = e;
    den += e;
  }
  float o[HDIM];
#pragma unroll
  for (int d = 0; d < HDIM; ++d) o[d] = 0.0f;
#pragma unroll
  for (int ky = 0; ky < 7; ++ky) {
#pragma unroll
    for (int kx = 0; kx < 7; ++kx) {
      const int yy = y + ky - 3;
      const int xx = x + kx - 3;
      if ((unsigned)yy < 64u && (unsigned)xx < 64u) {
        const float* vp = qkv + (size_t)(b * HW + yy * 64 + xx) * C3 + 2 * EMBED + h * HDIM;
        const float w = logit[ky * 7 + kx];
#pragma unroll
        for (int d = 0; d < HDIM; ++d) o[d] = fmaf(w, vp[d], o[d]);
      }
    }
  }
  const float inv = 1.0f / den;
  __bf16* op = attn_out + (size_t)(b * HW + y * 64 + x) * EMBED + h * HDIM;
#pragma unroll
  for (int d = 0; d < HDIM; ++d) op[d] = (__bf16)(o[d] * inv);
}

// ---------------------------------------------------------------------------
// Kernel 4: out = attn @ Wproj^T + bproj.  M=16384, N=128, K=128.
// grid(M/64, N/64), block 128; same LDS-staged panel scheme.
// ---------------------------------------------------------------------------
__global__ __launch_bounds__(128) void out_gemm(
    const __bf16* __restrict__ Aat, const __bf16* __restrict__ Wp,
    const float* __restrict__ bproj, float* __restrict__ out) {
  __shared__ __bf16 ldsB[64 * BSTRIDE];

  const int lane = threadIdx.x & 31;
  const int wave = threadIdx.x >> 5;
  const int lm   = lane & 15;
  const int half = lane >> 4;
  const int h8   = half * 8;
  const int row0 = blockIdx.x * 64 + wave * 16;
  const int col0 = blockIdx.y * 64;

  stage_b_panel(Wp + (size_t)col0 * EMBED, ldsB);

  const __bf16* arow = Aat + (size_t)(row0 + lm) * EMBED;

  v8f acc0 = {}, acc1 = {}, acc2 = {}, acc3 = {};
#pragma unroll
  for (int k0 = 0; k0 < EMBED; k0 += 32) {
    v16bf af;
    {
      const __bf16* ap = arow + k0 + h8;
#pragma unroll
      for (int e = 0; e < 8; ++e) {
        af[e]     = ap[e];
        af[e + 8] = ap[e + 16];
      }
    }
#pragma unroll
    for (int nt = 0; nt < 4; ++nt) {
      v16bf bfr;
      const __bf16* bp = ldsB + (nt * 16 + lm) * BSTRIDE + k0 + h8;
#pragma unroll
      for (int e = 0; e < 8; ++e) {
        bfr[e]     = bp[e];
        bfr[e + 8] = bp[e + 16];
      }
      v8f* accp = (nt == 0) ? &acc0 : (nt == 1) ? &acc1 : (nt == 2) ? &acc2 : &acc3;
      *accp = __builtin_amdgcn_wmma_f32_16x16x32_bf16(false, af, false, bfr,
                                                      (short)0, *accp, false, false);
    }
  }
#pragma unroll
  for (int nt = 0; nt < 4; ++nt) {
    const v8f acc = (nt == 0) ? acc0 : (nt == 1) ? acc1 : (nt == 2) ? acc2 : acc3;
    const int colT = col0 + nt * 16;
    const float bias = bproj[colT + lm];
    float* orow = out + (size_t)(row0 + 8 * half) * EMBED + colT + lm;
#pragma unroll
    for (int v = 0; v < 8; ++v) orow[(size_t)v * EMBED] = acc[v] + bias;
  }
}

// ---------------------------------------------------------------------------
// Workspace layout (bytes):
//   0        : c_proj  (32 f32)
//   128      : alpha   (4 f32)
//   256      : Weff    (196608 bf16 = 393216 B)
//   393472   : WprojBf (16384 bf16 = 32768 B)
//   426240   : qkv     (4*4096*384 f32 = 25165824 B)
//   25592064 : attnBf  (4*4096*128 bf16 = 4194304 B)
// ---------------------------------------------------------------------------
extern "C" void kernel_launch(void* const* d_in, const int* in_sizes, int n_in,
                              void* d_out, int out_size, void* d_ws, size_t ws_size,
                              hipStream_t stream) {
  (void)in_sizes; (void)n_in; (void)out_size; (void)ws_size;
  const float* x       = (const float*)d_in[0];
  const float* context = (const float*)d_in[1];
  const float* Wqkv    = (const float*)d_in[2];
  const float* bqkv    = (const float*)d_in[3];
  const float* A       = (const float*)d_in[4];
  const float* Blora   = (const float*)d_in[5];
  const float* Vlora   = (const float*)d_in[6];
  const float* g1w     = (const float*)d_in[7];
  const float* g1b     = (const float*)d_in[8];
  const float* g2w     = (const float*)d_in[9];
  const float* g2b     = (const float*)d_in[10];
  const float* Wproj   = (const float*)d_in[11];
  const float* bproj   = (const float*)d_in[12];
  float* out = (float*)d_out;

  char* ws = (char*)d_ws;
  float*  c_proj  = (float*)(ws + 0);
  float*  alpha   = (float*)(ws + 128);
  __bf16* Weff    = (__bf16*)(ws + 256);
  __bf16* WprojBf = (__bf16*)(ws + 393472);
  float*  qkv     = (float*)(ws + 426240);
  __bf16* attnBf  = (__bf16*)(ws + 25592064);

  prep_scalars<<<1, 64, 0, stream>>>(context, Blora, g1w, g1b, g2w, g2b, c_proj, alpha);

  const int prepTotal = WEFF_ELEMS + WPROJ_ELEMS;          // 212992
  prep_weights<<<(prepTotal + 255) / 256, 256, 0, stream>>>(
      Wqkv, A, Vlora, c_proj, alpha, Wproj, Weff, WprojBf);

  qkv_gemm<<<dim3(HW / 64, C3 / 64, BATCH), 128, 0, stream>>>(x, Weff, bqkv, qkv);

  neigh_attn<<<(BATCH * NHEADS * HW) / 256, 256, 0, stream>>>(qkv, attnBf);

  out_gemm<<<dim3((BATCH * HW) / 64, EMBED / 64, 1), 128, 0, stream>>>(
      attnBf, WprojBf, bproj, out);
}
